// Attention_39822936769178
// MI455X (gfx1250) — matrix operand order repaired and checked by reading, not compiled, not run on hardware
//
#include <hip/hip_runtime.h>
#include <hip/hip_bf16.h>
#include <stdint.h>

typedef __attribute__((ext_vector_type(16))) __bf16 v16bf;
typedef __attribute__((ext_vector_type(8)))  __bf16 v8bf;
typedef __attribute__((ext_vector_type(4)))  __bf16 v4bf;
typedef __attribute__((ext_vector_type(8)))  float  v8f;
typedef __attribute__((ext_vector_type(4)))  float  v4f;

#define D_MODEL 1024
#define NHEADS  16
#define DKH     64
#define BATCH   4
#define SEQ     2048
#define MTOT    (BATCH*SEQ)

// ---------------------------------------------------------------------------
// CDNA5 async global->LDS copy (tracked by ASYNCcnt, no VGPR round trip).
// lds_off = raw LDS byte offset (low 32 bits of the generic shared pointer).
// ---------------------------------------------------------------------------
__device__ __forceinline__ void async_load_b128(unsigned lds_off, const void* gaddr) {
  asm volatile("global_load_async_to_lds_b128 %0, %1, off"
               :: "v"(lds_off), "v"((unsigned long long)(size_t)gaddr)
               : "memory");
}
__device__ __forceinline__ void wait_async0() {
  asm volatile("s_wait_asynccnt 0x0" ::: "memory");
}
__device__ __forceinline__ unsigned lds_off32(const void* p) {
  return (unsigned)(size_t)p;   // LDS aperture: addr[31:0] == LDS byte offset
}

// ---------------------------------------------------------------------------
// fp32 -> bf16 cast, 4 elements/thread, packed 8B store
// ---------------------------------------------------------------------------
__global__ __launch_bounds__(256)
void cast_f32_bf16_x4(const float* __restrict__ src, __bf16* __restrict__ dst, int n4) {
  int i = blockIdx.x * blockDim.x + threadIdx.x;
  if (i < n4) {
    v4f v = *(const v4f*)(src + 4 * (size_t)i);
    v4bf o;
    o[0] = (__bf16)v.x; o[1] = (__bf16)v.y; o[2] = (__bf16)v.z; o[3] = (__bf16)v.w;
    *(v4bf*)(dst + 4 * (size_t)i) = o;
  }
}

// ---------------------------------------------------------------------------
// C[m,n] = sum_k A[m,k] * W[n,k] + bias[n]   (A @ W^T + b)
// Double-buffered LDS tiles filled by async-to-LDS DMA, overlapped with WMMA.
// mode 0: bf16 out with head permute [B][H][L][64];  mode 1: fp32 out[m*N+n]
// block = 128 threads (4 waves, 2x2), block tile 128x128, k-step 32
// ---------------------------------------------------------------------------
__global__ __launch_bounds__(128)
void gemm_wmma_bf16(const __bf16* __restrict__ A,
                    const __bf16* __restrict__ W,
                    const float*  __restrict__ bias,
                    void* __restrict__ out,
                    int M, int N, int K, int mode, int Lseq)
{
  // rows padded to 48 elems (96 B) so v16bf(32B)/v8bf(16B) LDS loads stay aligned
  __shared__ __bf16 As[2][128][48];
  __shared__ __bf16 Ws[2][128][48];

  const int tid  = threadIdx.x;
  const int wave = tid >> 5;
  const int lane = tid & 31;
  const int waveM = (wave >> 1) * 64;
  const int waveN = (wave & 1)  * 64;
  const int blockM = blockIdx.x * 128;
  const int blockN = blockIdx.y * 128;
  const int mrow = lane & 15;
  const int k8   = (lane >> 4) * 8;
  const int k16  = (lane >> 4) * 16;
  const int hi8  = (lane >> 4) * 8;

  v8f acc[4][4] = {};

  const __bf16* Ag = A + (size_t)(blockM + tid) * K;
  const __bf16* Wg = W + (size_t)(blockN + tid) * K;
  const unsigned ldsA0 = lds_off32(&As[0][tid][0]);
  const unsigned ldsA1 = lds_off32(&As[1][tid][0]);
  const unsigned ldsW0 = lds_off32(&Ws[0][tid][0]);
  const unsigned ldsW1 = lds_off32(&Ws[1][tid][0]);

  auto issue_tile = [&](int kt, int b) {
    unsigned la = b ? ldsA1 : ldsA0;
    unsigned lw = b ? ldsW1 : ldsW0;
    #pragma unroll
    for (int c = 0; c < 32; c += 8) {            // 4x b128 per row, per matrix
      async_load_b128(la + 2 * c, &Ag[kt + c]);
      async_load_b128(lw + 2 * c, &Wg[kt + c]);
    }
  };

  issue_tile(0, 0);
  int it = 0;
  for (int kt = 0; kt < K; kt += 32, ++it) {
    const int b = it & 1;
    wait_async0();           // my async writes into buffer b have landed
    __syncthreads();         // everyone's writes have landed; prev reads done
    if (kt + 32 < K) issue_tile(kt + 32, b ^ 1);   // DMA next tile during compute

    v16bf af[4], wf[4];
    #pragma unroll
    for (int i = 0; i < 4; ++i) {
      const __bf16* p = &As[b][waveM + i * 16 + mrow][0];
      v8bf lo = *(const v8bf*)&p[k16];
      v8bf hi = *(const v8bf*)&p[k16 + 8];
      #pragma unroll
      for (int e = 0; e < 8; ++e) { af[i][e] = lo[e]; af[i][8 + e] = hi[e]; }
    }
    #pragma unroll
    for (int j = 0; j < 4; ++j)
      wf[j] = *(const v16bf*)&Ws[b][waveN + j * 16 + mrow][k16];

    #pragma unroll
    for (int i = 0; i < 4; ++i)
      #pragma unroll
      for (int j = 0; j < 4; ++j)
        acc[i][j] = __builtin_amdgcn_wmma_f32_16x16x32_bf16(
            false, af[i], false, wf[j], (short)0, acc[i][j], false, false);
  }

  // epilogue: C layout -> VGPR r holds (M = r + hi8, N = lane&15)
  #pragma unroll
  for (int i = 0; i < 4; ++i)
    #pragma unroll
    for (int j = 0; j < 4; ++j)
      #pragma unroll
      for (int r = 0; r < 8; ++r) {
        int m = blockM + waveM + i * 16 + r + hi8;
        int n = blockN + waveN + j * 16 + mrow;
        float v = acc[i][j][r] + bias[n];
        if (mode == 0) {
          int bb = m / Lseq, l = m - bb * Lseq;
          int h = n >> 6, dk = n & 63;
          ((__bf16*)out)[(((size_t)(bb * NHEADS + h) * Lseq + l) << 6) + dk] = (__bf16)v;
        } else {
          ((float*)out)[(size_t)m * N + n] = v;
        }
      }
}

// ---------------------------------------------------------------------------
// Flash attention (causal), bf16 WMMA, fp32 softmax statistics.
// grid.x = B*H, grid.y = L/64. Block = 128 threads = 4 waves; each wave owns
// 16 query rows. K tile DMA'd to LDS (async); V tile staged via registers and
// stored transposed. q/k/v: [B,H,L,64] bf16; output: [B,L,D_MODEL] bf16.
// ---------------------------------------------------------------------------
__global__ __launch_bounds__(128)
void flash_attn_wmma(const __bf16* __restrict__ qh, const __bf16* __restrict__ kh,
                     const __bf16* __restrict__ vh, __bf16* __restrict__ attn_out,
                     int Lseq)
{
  __shared__ __bf16 Ks[64][80];        // [key][dim]   row stride 160 B
  __shared__ __bf16 Vt[64][80];        // [dim][key]   transposed
  __shared__ __bf16 Ps[4][16][80];     // per-wave P scratch [row][key]

  const int bh = blockIdx.x;
  const int qt = blockIdx.y;
  const int q0 = qt * 64;
  const int tid  = threadIdx.x;
  const int wave = tid >> 5;
  const int lane = tid & 31;
  const int mrow = lane & 15;
  const int k8   = (lane >> 4) * 8;
  const int k16  = (lane >> 4) * 16;
  const int hi8  = (lane >> 4) * 8;
  const size_t baseBH = (size_t)bh * Lseq * DKH;

  // Q fragments for this wave's 16 rows, pre-scaled by 1/sqrt(64)
  v16bf qf[2];
  {
    const __bf16* qp = qh + baseBH + (size_t)(q0 + wave * 16 + mrow) * DKH;
    #pragma unroll
    for (int d = 0; d < 2; ++d) {
      v8bf lo = *(const v8bf*)&qp[d * 32 + k16];
      v8bf hi = *(const v8bf*)&qp[d * 32 + k16 + 8];
      #pragma unroll
      for (int e = 0; e < 8; ++e) {
        qf[d][e]     = (__bf16)((float)lo[e] * 0.125f);
        qf[d][8 + e] = (__bf16)((float)hi[e] * 0.125f);
      }
    }
  }

  v8f o[4] = {};
  float mst[8], lst[8];
  #pragma unroll
  for (int r = 0; r < 8; ++r) { mst[r] = -3.0e38f; lst[r] = 0.f; }

  const int row = tid >> 1, half = (tid & 1) * 32;   // cooperative-load mapping
  const unsigned ldsK = lds_off32(&Ks[row][half]);

  for (int kt = 0; kt <= qt; ++kt) {
    const int k0 = kt * 64;
    {
      const __bf16* kg = kh + baseBH + (size_t)(k0 + row) * DKH + half;
      const __bf16* vg = vh + baseBH + (size_t)(k0 + row) * DKH + half;
      async_load_b128(ldsK,      kg);          // K tile: DMA straight to LDS
      async_load_b128(ldsK + 16, kg + 8);
      async_load_b128(ldsK + 32, kg + 16);
      async_load_b128(ldsK + 48, kg + 24);
      v8bf v0 = *(const v8bf*)&vg[0];          // V tile: 4 loads in flight,
      v8bf v1 = *(const v8bf*)&vg[8];          // then transpose-store
      v8bf v2 = *(const v8bf*)&vg[16];
      v8bf v3 = *(const v8bf*)&vg[24];
      #pragma unroll
      for (int e = 0; e < 8; ++e) {
        Vt[half + e][row]      = v0[e];
        Vt[half + 8 + e][row]  = v1[e];
        Vt[half + 16 + e][row] = v2[e];
        Vt[half + 24 + e][row] = v3[e];
      }
    }
    wait_async0();
    __syncthreads();

    // S = Q K^T
    v8f s[4] = {};
    #pragma unroll
    for (int j = 0; j < 4; ++j)
      #pragma unroll
      for (int d = 0; d < 2; ++d) {
        v16bf kf = *(const v16bf*)&Ks[j * 16 + mrow][d * 32 + k16];
        s[j] = __builtin_amdgcn_wmma_f32_16x16x32_bf16(
            false, qf[d], false, kf, (short)0, s[j], false, false);
      }

    if (kt == qt) {                 // causal mask on the diagonal tile
      #pragma unroll
      for (int j = 0; j < 4; ++j)
        #pragma unroll
        for (int r = 0; r < 8; ++r)
          if (j * 16 + mrow > wave * 16 + r + hi8) s[j][r] = -3.0e38f;
    }

    // online softmax update (row reductions across 16-lane halves)
    float sc[8];
    #pragma unroll
    for (int r = 0; r < 8; ++r) {
      float v = fmaxf(fmaxf(s[0][r], s[1][r]), fmaxf(s[2][r], s[3][r]));
      v = fmaxf(v, __shfl_xor(v, 1, 16));
      v = fmaxf(v, __shfl_xor(v, 2, 16));
      v = fmaxf(v, __shfl_xor(v, 4, 16));
      v = fmaxf(v, __shfl_xor(v, 8, 16));
      float mnew = fmaxf(mst[r], v);
      sc[r] = __expf(mst[r] - mnew);
      mst[r] = mnew;
      float acc = 0.f;
      #pragma unroll
      for (int j = 0; j < 4; ++j) { float p = __expf(s[j][r] - mnew); s[j][r] = p; acc += p; }
      acc += __shfl_xor(acc, 1, 16);
      acc += __shfl_xor(acc, 2, 16);
      acc += __shfl_xor(acc, 4, 16);
      acc += __shfl_xor(acc, 8, 16);
      lst[r] = lst[r] * sc[r] + acc;
    }
    #pragma unroll
    for (int j = 0; j < 4; ++j)
      #pragma unroll
      for (int r = 0; r < 8; ++r) o[j][r] *= sc[r];

    // P (C layout) -> per-wave LDS scratch for A-frag reload
    #pragma unroll
    for (int j = 0; j < 4; ++j)
      #pragma unroll
      for (int r = 0; r < 8; ++r)
        Ps[wave][r + hi8][j * 16 + mrow] = (__bf16)s[j][r];

    // O += P @ V
    v16bf pf[2];
    #pragma unroll
    for (int kc = 0; kc < 2; ++kc) {
      const __bf16* pp = &Ps[wave][mrow][kc * 32];
      v8bf lo = *(const v8bf*)&pp[k16];
      v8bf hi = *(const v8bf*)&pp[k16 + 8];
      #pragma unroll
      for (int e = 0; e < 8; ++e) { pf[kc][e] = lo[e]; pf[kc][8 + e] = hi[e]; }
    }
    #pragma unroll
    for (int j = 0; j < 4; ++j)
      #pragma unroll
      for (int kc = 0; kc < 2; ++kc) {
        v16bf vf = *(const v16bf*)&Vt[j * 16 + mrow][kc * 32 + k16];
        o[j] = __builtin_amdgcn_wmma_f32_16x16x32_bf16(
            false, pf[kc], false, vf, (short)0, o[j], false, false);
      }
    __syncthreads();
  }

  // normalize and write [B, L, D_MODEL] bf16
  const int h = bh & (NHEADS - 1), b = bh >> 4;
  #pragma unroll
  for (int r = 0; r < 8; ++r) {
    float inv = 1.0f / lst[r];
    int qrow = q0 + wave * 16 + r + hi8;
    #pragma unroll
    for (int j = 0; j < 4; ++j) {
      int d = j * 16 + mrow;
      attn_out[((size_t)(b * Lseq + qrow) << 10) + h * DKH + d] = (__bf16)(o[j][r] * inv);
    }
  }
}

// ---------------------------------------------------------------------------
extern "C" void kernel_launch(void* const* d_in, const int* in_sizes, int n_in,
                              void* d_out, int out_size, void* d_ws, size_t ws_size,
                              hipStream_t stream) {
  (void)in_sizes; (void)n_in; (void)out_size; (void)ws_size;
  const float* Q  = (const float*)d_in[0];
  const float* K  = (const float*)d_in[1];
  const float* V  = (const float*)d_in[2];
  // d_in[3] = attn_mask (causal, handled analytically)
  const float* Wq = (const float*)d_in[4];
  const float* bq = (const float*)d_in[5];
  const float* Wk = (const float*)d_in[6];
  const float* bk = (const float*)d_in[7];
  const float* Wv = (const float*)d_in[8];
  const float* bv = (const float*)d_in[9];
  const float* Wo = (const float*)d_in[10];
  const float* bo = (const float*)d_in[11];
  float* out = (float*)d_out;

  char* w = (char*)d_ws;
  const size_t MB = 1ull << 20;
  __bf16* Qb   = (__bf16*)(w + 0   * MB);
  __bf16* Kb   = (__bf16*)(w + 16  * MB);
  __bf16* Vb   = (__bf16*)(w + 32  * MB);
  __bf16* Wqb  = (__bf16*)(w + 48  * MB);
  __bf16* Wkb  = (__bf16*)(w + 50  * MB);
  __bf16* Wvb  = (__bf16*)(w + 52  * MB);
  __bf16* Wob  = (__bf16*)(w + 54  * MB);
  __bf16* qh   = (__bf16*)(w + 56  * MB);
  __bf16* kh   = (__bf16*)(w + 72  * MB);
  __bf16* vh   = (__bf16*)(w + 88  * MB);
  __bf16* attn = (__bf16*)(w + 104 * MB);

  const int nQKV = MTOT * D_MODEL;
  const int nW   = D_MODEL * D_MODEL;
  cast_f32_bf16_x4<<<nQKV / 4 / 256, 256, 0, stream>>>(Q,  Qb,  nQKV / 4);
  cast_f32_bf16_x4<<<nQKV / 4 / 256, 256, 0, stream>>>(K,  Kb,  nQKV / 4);
  cast_f32_bf16_x4<<<nQKV / 4 / 256, 256, 0, stream>>>(V,  Vb,  nQKV / 4);
  cast_f32_bf16_x4<<<nW   / 4 / 256, 256, 0, stream>>>(Wq, Wqb, nW / 4);
  cast_f32_bf16_x4<<<nW   / 4 / 256, 256, 0, stream>>>(Wk, Wkb, nW / 4);
  cast_f32_bf16_x4<<<nW   / 4 / 256, 256, 0, stream>>>(Wv, Wvb, nW / 4);
  cast_f32_bf16_x4<<<nW   / 4 / 256, 256, 0, stream>>>(Wo, Wob, nW / 4);

  dim3 g(MTOT / 128, D_MODEL / 128);
  gemm_wmma_bf16<<<g, 128, 0, stream>>>(Qb, Wqb, bq, qh, MTOT, D_MODEL, D_MODEL, 0, SEQ);
  gemm_wmma_bf16<<<g, 128, 0, stream>>>(Kb, Wkb, bk, kh, MTOT, D_MODEL, D_MODEL, 0, SEQ);
  gemm_wmma_bf16<<<g, 128, 0, stream>>>(Vb, Wvb, bv, vh, MTOT, D_MODEL, D_MODEL, 0, SEQ);

  flash_attn_wmma<<<dim3(BATCH * NHEADS, SEQ / 64), 128, 0, stream>>>(qh, kh, vh, attn, SEQ);

  gemm_wmma_bf16<<<g, 128, 0, stream>>>(attn, Wob, bo, out, MTOT, D_MODEL, D_MODEL, 1, SEQ);
}
